// GCNII_Encoder_59425167507606
// MI455X (gfx1250) — matrix operand order repaired and checked by reading, not compiled, not run on hardware
//
#include <hip/hip_runtime.h>
#include <math.h>

// ---- problem constants (match reference) ----
#define N_NODES 50000
#define E_EDGES 800000
#define DIN     500
#define KPAD1   512     // DIN padded to multiple of 32
#define H       128
#define NCLS    40
#define BSZ     10000
#define NLAY    9
// ALPHA=0.5 -> (1-ALPHA)=0.5 folded into spmm; xr = 0.5*x0
// ETA=0.5

typedef __attribute__((ext_vector_type(16))) __bf16 v16bf;
typedef __attribute__((ext_vector_type(8)))  float  v8f;

__device__ __forceinline__ unsigned short f32_to_bf16(float f) {
  unsigned int u = __float_as_uint(f);
  u += 0x7FFFu + ((u >> 16) & 1u);   // round-to-nearest-even
  return (unsigned short)(u >> 16);
}

// ---------------------------------------------------------------------------
// Pack a row-major MxKsrc fp32 matrix into bf16 WMMA A-fragments (16x32 tiles).
// Fragment order: [rowtile][ktile][lane*16 + j], lane layout (contiguous k, same order as the B packers):
//   m = lane&15 ; half = lane>>4 ; k = kt*32 + half*16 + j
// ---------------------------------------------------------------------------
__global__ void pack_a_kernel(const float* __restrict__ src,
                              unsigned short* __restrict__ dst,
                              int M, int Ksrc, int Kpad) {
  long long total = (long long)(M >> 4) * (Kpad >> 5) * 512;
  long long tid = (long long)blockIdx.x * blockDim.x + threadIdx.x;
  if (tid >= total) return;
  int within   = (int)(tid & 511);
  long long tl = tid >> 9;
  int ktiles   = Kpad >> 5;
  int rowtile  = (int)(tl / ktiles);
  int kt       = (int)(tl % ktiles);
  int lane = within >> 4, j = within & 15;
  int half = lane >> 4;
  int m = rowtile * 16 + (lane & 15);
  int kl = half * 16 + j;
  int k = kt * 32 + kl;
  float v = (k < Ksrc) ? src[(long long)m * Ksrc + k] : 0.0f;
  dst[tid] = f32_to_bf16(v);
}

// ---------------------------------------------------------------------------
// Pack a row-major Ksrc x Nsrc fp32 matrix into bf16 WMMA B-fragments (32x16).
// Fragment order: [ntile][ktile][lane*16 + j]:
//   n = lane&15 ; half = lane>>4 ; k = kt*32 + half*16 + j
// ---------------------------------------------------------------------------
__global__ void pack_b_kernel(const float* __restrict__ src,
                              unsigned short* __restrict__ dst,
                              int Ksrc, int Nsrc, int Kpad, int Npad) {
  long long total = (long long)(Npad >> 4) * (Kpad >> 5) * 512;
  long long tid = (long long)blockIdx.x * blockDim.x + threadIdx.x;
  if (tid >= total) return;
  int within   = (int)(tid & 511);
  long long tl = tid >> 9;
  int ktiles   = Kpad >> 5;
  int nt = (int)(tl / ktiles);
  int kt = (int)(tl % ktiles);
  int lane = within >> 4, j = within & 15;
  int half = lane >> 4;
  int n = nt * 16 + (lane & 15);
  int k = kt * 32 + half * 16 + j;
  float v = (k < Ksrc && n < Nsrc) ? src[(long long)k * Nsrc + n] : 0.0f;
  dst[tid] = f32_to_bf16(v);
}

// B-fragments of src^T where src is Msrc x Ksrc row-major (for sim = an @ an^T)
__global__ void pack_bt_kernel(const float* __restrict__ src,
                               unsigned short* __restrict__ dst,
                               int Msrc, int Ksrc) {
  long long total = (long long)(Msrc >> 4) * (Ksrc >> 5) * 512;
  long long tid = (long long)blockIdx.x * blockDim.x + threadIdx.x;
  if (tid >= total) return;
  int within   = (int)(tid & 511);
  long long tl = tid >> 9;
  int ktiles   = Ksrc >> 5;
  int nt = (int)(tl / ktiles);
  int kt = (int)(tl % ktiles);
  int lane = within >> 4, j = within & 15;
  int half = lane >> 4;
  int n = nt * 16 + (lane & 15);              // row of src == column of src^T
  int k = kt * 32 + half * 16 + j;
  float v = src[(long long)n * Ksrc + k];
  dst[tid] = f32_to_bf16(v);
}

// ---------------------------------------------------------------------------
// GEMM1: h = relu(x @ W1 + b1); also xr = 0.5*h (= ALPHA * x0, loop invariant)
// One block per 16-row tile; 8 waves = 8 column tiles of H=128. K = 512 (16 steps).
// ---------------------------------------------------------------------------
__global__ void gemm1_kernel(const unsigned short* __restrict__ xp,
                             const unsigned short* __restrict__ w1p,
                             const float* __restrict__ b1,
                             float* __restrict__ h, float* __restrict__ xr) {
  int rowtile = blockIdx.x;
  int wave = threadIdx.x >> 5, lane = threadIdx.x & 31;
  v8f acc = {};
  const v16bf* A = (const v16bf*)xp;
  const v16bf* Bm = (const v16bf*)w1p;
#pragma unroll 4
  for (int kt = 0; kt < (KPAD1 >> 5); ++kt) {
    v16bf a = A[(size_t)(rowtile * (KPAD1 >> 5) + kt) * 32 + lane];
    v16bf b = Bm[(size_t)(wave * (KPAD1 >> 5) + kt) * 32 + lane];
    acc = __builtin_amdgcn_wmma_f32_16x16x32_bf16(false, a, false, b,
                                                  (short)0, acc, false, false);
  }
  int half = lane >> 4;
  int n = wave * 16 + (lane & 15);
  float bias = b1[n];
#pragma unroll
  for (int v = 0; v < 8; ++v) {
    int m = v + 8 * half;
    long long r = (long long)rowtile * 16 + m;
    float hn = fmaxf(acc[v] + bias, 0.0f);
    h[r * H + n]  = hn;
    xr[r * H + n] = 0.5f * hn;
  }
}

// ---------------------------------------------------------------------------
// SpMM: p[row] += (1-ALPHA) * w_e * h[col]   (fp32 atomics; bandwidth bound)
// One thread handles 4 channels of one edge (float4 gather, 4 atomic adds).
// ---------------------------------------------------------------------------
__global__ void spmm_kernel(const int* __restrict__ ei,
                            const float* __restrict__ ew,
                            const float* __restrict__ h,
                            float* __restrict__ p) {
  long long tid = (long long)blockIdx.x * blockDim.x + threadIdx.x;
  if (tid >= (long long)E_EDGES * 32) return;
  int e = (int)(tid >> 5), g = (int)(tid & 31);
  int row = ei[e];
  int col = ei[E_EDGES + e];
  float w = ew[e] * 0.5f;                       // (1-ALPHA)
  float4 hv = ((const float4*)(h + (long long)col * H))[g];
  float* dst = p + (long long)row * H + g * 4;
  atomicAdd(dst + 0, w * hv.x);
  atomicAdd(dst + 1, w * hv.y);
  atomicAdd(dst + 2, w * hv.z);
  atomicAdd(dst + 3, w * hv.w);
}

// ---------------------------------------------------------------------------
// Fused layer: out = (1-b)*(p+xr) + b*(p@w1 + xr@w2); h = relu(h + out), in place.
// One block per 16-row tile, wave -> column tile. K = 128 (4 WMMA steps x 2 GEMMs).
// ---------------------------------------------------------------------------
__global__ void layer_kernel(const unsigned short* __restrict__ pp,
                             const unsigned short* __restrict__ xrp,
                             const unsigned short* __restrict__ w1f,
                             const unsigned short* __restrict__ w2f,
                             const float* __restrict__ p,
                             const float* __restrict__ xr,
                             float* __restrict__ h, float beta) {
  int rowtile = blockIdx.x;
  int wave = threadIdx.x >> 5, lane = threadIdx.x & 31;
  v8f acc1 = {}, acc2 = {};
  const v16bf* Pp = (const v16bf*)pp;
  const v16bf* Xp = (const v16bf*)xrp;
  const v16bf* B1 = (const v16bf*)w1f;
  const v16bf* B2 = (const v16bf*)w2f;
#pragma unroll
  for (int kt = 0; kt < 4; ++kt) {
    v16bf ap = Pp[(size_t)(rowtile * 4 + kt) * 32 + lane];
    v16bf b1 = B1[(size_t)(wave * 4 + kt) * 32 + lane];
    acc1 = __builtin_amdgcn_wmma_f32_16x16x32_bf16(false, ap, false, b1,
                                                   (short)0, acc1, false, false);
    v16bf ax = Xp[(size_t)(rowtile * 4 + kt) * 32 + lane];
    v16bf b2 = B2[(size_t)(wave * 4 + kt) * 32 + lane];
    acc2 = __builtin_amdgcn_wmma_f32_16x16x32_bf16(false, ax, false, b2,
                                                   (short)0, acc2, false, false);
  }
  int half = lane >> 4;
  int n = wave * 16 + (lane & 15);
  float omb = 1.0f - beta;
#pragma unroll
  for (int v = 0; v < 8; ++v) {
    int m = v + 8 * half;
    long long idx = ((long long)rowtile * 16 + m) * H + n;
    float pv = p[idx], xv = xr[idx], hv = h[idx];
    float out = omb * (pv + xv) + beta * (acc1[v] + acc2[v]);
    h[idx] = fmaxf(hv + out, 0.0f);
  }
}

// ---------------------------------------------------------------------------
// logits[:B] = h[:B] @ W2 + b2 ; p_lc = log_softmax  (tiny: 102 MFLOP, VALU ok)
// ---------------------------------------------------------------------------
__global__ void logits_kernel(const float* __restrict__ h,
                              const float* __restrict__ W2,
                              const float* __restrict__ b2,
                              float* __restrict__ plc) {
  int r = blockIdx.x * blockDim.x + threadIdx.x;
  if (r >= BSZ) return;
  float acc[NCLS];
#pragma unroll
  for (int c = 0; c < NCLS; ++c) acc[c] = b2[c];
  for (int k = 0; k < H; ++k) {
    float hv = h[(long long)r * H + k];
#pragma unroll
    for (int c = 0; c < NCLS; ++c) acc[c] += hv * W2[k * NCLS + c];
  }
  float mx = acc[0];
#pragma unroll
  for (int c = 1; c < NCLS; ++c) mx = fmaxf(mx, acc[c]);
  float s = 0.0f;
#pragma unroll
  for (int c = 0; c < NCLS; ++c) s += __expf(acc[c] - mx);
  float lse = mx + __logf(s);
#pragma unroll
  for (int c = 0; c < NCLS; ++c) plc[(long long)r * NCLS + c] = acc[c] - lse;
}

// normalize rows of h[:B] -> an ; also emit emb (= raw h[:B]) to output
__global__ void norm_emb_kernel(const float* __restrict__ h,
                                float* __restrict__ an,
                                float* __restrict__ emb_out) {
  int r = blockIdx.x * blockDim.x + threadIdx.x;
  if (r >= BSZ) return;
  float s = 0.0f;
  for (int k = 0; k < H; ++k) { float v = h[(long long)r * H + k]; s += v * v; }
  float inv = 1.0f / fmaxf(sqrtf(s), 1e-8f);
  for (int k = 0; k < H; ++k) {
    float v = h[(long long)r * H + k];
    an[(long long)r * H + k] = v * inv;
    emb_out[(long long)r * H + k] = v;
  }
}

// ---------------------------------------------------------------------------
// Fused sim = an@an^T with streaming top-5: never materializes the 400MB sim.
// One wave per 16-row tile; loops over 625 column tiles; 4 WMMAs per tile;
// 16x16 result bounced through LDS; lanes 0..15 each own one row's top-5.
// ---------------------------------------------------------------------------
__global__ void sim_topk_kernel(const unsigned short* __restrict__ aA,
                                const unsigned short* __restrict__ aB,
                                float* __restrict__ tkv, int* __restrict__ tki) {
  __shared__ float tile[256];
  int rowtile = blockIdx.x;
  int lane = threadIdx.x & 31;
  const v16bf* A = (const v16bf*)aA;
  const v16bf* Bm = (const v16bf*)aB;
  v16bf afr[4];
#pragma unroll
  for (int kt = 0; kt < 4; ++kt)
    afr[kt] = A[(size_t)(rowtile * 4 + kt) * 32 + lane];

  float tv[5]; int ti[5];
#pragma unroll
  for (int q = 0; q < 5; ++q) { tv[q] = -1e30f; ti[q] = 0; }

  int half = lane >> 4, nl = lane & 15;
  for (int jt = 0; jt < (BSZ >> 4); ++jt) {
    v8f acc = {};
#pragma unroll
    for (int kt = 0; kt < 4; ++kt) {
      v16bf b = Bm[(size_t)(jt * 4 + kt) * 32 + lane];
      acc = __builtin_amdgcn_wmma_f32_16x16x32_bf16(false, afr[kt], false, b,
                                                    (short)0, acc, false, false);
    }
#pragma unroll
    for (int v = 0; v < 8; ++v) tile[(v + 8 * half) * 16 + nl] = acc[v];
    __syncthreads();
    if (lane < 16) {
      for (int n2 = 0; n2 < 16; ++n2) {
        float s = tile[lane * 16 + n2];
        if (s > tv[4]) {
          tv[4] = s; ti[4] = jt * 16 + n2;
#pragma unroll
          for (int q = 4; q > 0; --q) {
            if (tv[q] > tv[q - 1]) {
              float tf = tv[q]; tv[q] = tv[q - 1]; tv[q - 1] = tf;
              int t2 = ti[q]; ti[q] = ti[q - 1]; ti[q - 1] = t2;
            }
          }
        }
      }
    }
    __syncthreads();
  }
  if (lane < 16) {
    long long r = (long long)rowtile * 16 + lane;
#pragma unroll
    for (int q = 0; q < 5; ++q) { tkv[r * 5 + q] = tv[q]; tki[r * 5 + q] = ti[q]; }
  }
}

// fused = sum_k exp(vals_k) * onehot(y[idx_k]); final = .5*p_lc + .5*logsoftmax(fused)
__global__ void finalize_kernel(const float* __restrict__ tkv,
                                const int* __restrict__ tki,
                                const int* __restrict__ y,
                                const float* __restrict__ plc,
                                float* __restrict__ out_final) {
  int r = blockIdx.x * blockDim.x + threadIdx.x;
  if (r >= BSZ) return;
  float fused[NCLS];
#pragma unroll
  for (int c = 0; c < NCLS; ++c) fused[c] = 0.0f;
#pragma unroll
  for (int k = 0; k < 5; ++k) {
    float w = __expf(tkv[(long long)r * 5 + k]);
    int cls = y[tki[(long long)r * 5 + k]];
    fused[cls] += w;
  }
  float mx = fused[0];
#pragma unroll
  for (int c = 1; c < NCLS; ++c) mx = fmaxf(mx, fused[c]);
  float s = 0.0f;
#pragma unroll
  for (int c = 0; c < NCLS; ++c) s += __expf(fused[c] - mx);
  float lse = mx + __logf(s);
#pragma unroll
  for (int c = 0; c < NCLS; ++c)
    out_final[(long long)r * NCLS + c] =
        0.5f * plc[(long long)r * NCLS + c] + 0.5f * (fused[c] - lse);
}

// ---------------------------------------------------------------------------
extern "C" void kernel_launch(void* const* d_in, const int* in_sizes, int n_in,
                              void* d_out, int out_size, void* d_ws, size_t ws_size,
                              hipStream_t stream) {
  (void)in_sizes; (void)n_in; (void)out_size; (void)ws_size;
  const float* x   = (const float*)d_in[0];
  const int*   ei  = (const int*)d_in[1];
  const float* ew  = (const float*)d_in[2];
  const int*   y   = (const int*)d_in[3];
  const float* W1  = (const float*)d_in[5];
  const float* b1  = (const float*)d_in[6];
  const float* cw1 = (const float*)d_in[7];
  const float* cw2 = (const float*)d_in[8];
  const float* W2  = (const float*)d_in[9];
  const float* b2  = (const float*)d_in[10];

  float* out_final = (float*)d_out;                    // BSZ*NCLS
  float* out_emb   = (float*)d_out + (size_t)BSZ * NCLS; // BSZ*H

  // ---- workspace layout ----
  char* ws = (char*)d_ws;
  size_t off = 0;
  auto alloc = [&](size_t bytes) {
    void* ptr = ws + off;
    off = (off + bytes + 255) & ~(size_t)255;
    return ptr;
  };
  float* h      = (float*)alloc((size_t)N_NODES * H * 4);
  float* xr     = (float*)alloc((size_t)N_NODES * H * 4);
  float* p      = (float*)alloc((size_t)N_NODES * H * 4);
  unsigned short* ppack  = (unsigned short*)alloc((size_t)N_NODES * H * 2);
  unsigned short* xrpack = (unsigned short*)alloc((size_t)N_NODES * H * 2);
  unsigned short* xpack  = (unsigned short*)alloc((size_t)N_NODES * KPAD1 * 2);
  unsigned short* w1pack = (unsigned short*)alloc((size_t)KPAD1 * H * 2);
  unsigned short* cwpack = (unsigned short*)alloc((size_t)2 * NLAY * H * H * 2);
  float* plc    = (float*)alloc((size_t)BSZ * NCLS * 4);
  float* an     = (float*)alloc((size_t)BSZ * H * 4);
  unsigned short* anA = (unsigned short*)alloc((size_t)BSZ * H * 2);
  unsigned short* anB = (unsigned short*)alloc((size_t)BSZ * H * 2);
  float* tkv    = (float*)alloc((size_t)BSZ * 5 * 4);
  int*   tki    = (int*)alloc((size_t)BSZ * 5 * 4);

  const int TB = 256;

  // pack x (pad K 500->512) and W1
  {
    long long n = (long long)(N_NODES / 16) * (KPAD1 / 32) * 512;
    pack_a_kernel<<<(int)((n + TB - 1) / TB), TB, 0, stream>>>(x, xpack, N_NODES, DIN, KPAD1);
  }
  {
    long long n = (long long)(H / 16) * (KPAD1 / 32) * 512;
    pack_b_kernel<<<(int)((n + TB - 1) / TB), TB, 0, stream>>>(W1, w1pack, DIN, H, KPAD1, H);
  }
  // pack 9x conv_w1 and 9x conv_w2 (each 128x128 -> 32KB bf16 fragments)
  {
    long long n = (long long)(H / 16) * (H / 32) * 512;
    int g = (int)((n + TB - 1) / TB);
    for (int l = 0; l < NLAY; ++l) {
      pack_b_kernel<<<g, TB, 0, stream>>>(cw1 + (size_t)l * H * H,
                                          cwpack + (size_t)l * H * H, H, H, H, H);
      pack_b_kernel<<<g, TB, 0, stream>>>(cw2 + (size_t)l * H * H,
                                          cwpack + (size_t)(NLAY + l) * H * H, H, H, H, H);
    }
  }
  // h = relu(x@W1+b1) ; xr = 0.5*h
  gemm1_kernel<<<N_NODES / 16, 256, 0, stream>>>(xpack, w1pack, b1, h, xr);
  // pack xr once (loop invariant)
  {
    long long n = (long long)(N_NODES / 16) * (H / 32) * 512;
    pack_a_kernel<<<(int)((n + TB - 1) / TB), TB, 0, stream>>>(xr, xrpack, N_NODES, H, H);
  }

  long long nsp = (long long)E_EDGES * 32;
  long long npk = (long long)(N_NODES / 16) * (H / 32) * 512;
  for (int l = 0; l < NLAY; ++l) {
    hipMemsetAsync(p, 0, (size_t)N_NODES * H * 4, stream);
    spmm_kernel<<<(int)((nsp + TB - 1) / TB), TB, 0, stream>>>(ei, ew, h, p);
    pack_a_kernel<<<(int)((npk + TB - 1) / TB), TB, 0, stream>>>(p, ppack, N_NODES, H, H);
    float beta = logf(1.0f / (float)(l + 1) + 1.0f);   // THETA = 1.0
    layer_kernel<<<N_NODES / 16, 256, 0, stream>>>(
        ppack, xrpack,
        cwpack + (size_t)l * H * H,
        cwpack + (size_t)(NLAY + l) * H * H,
        p, xr, h, beta);
  }

  logits_kernel<<<(BSZ + TB - 1) / TB, TB, 0, stream>>>(h, W2, b2, plc);
  norm_emb_kernel<<<(BSZ + TB - 1) / TB, TB, 0, stream>>>(h, an, out_emb);
  {
    long long n = (long long)(BSZ / 16) * (H / 32) * 512;
    pack_a_kernel<<<(int)((n + TB - 1) / TB), TB, 0, stream>>>(an, anA, BSZ, H, H);
    pack_bt_kernel<<<(int)((n + TB - 1) / TB), TB, 0, stream>>>(an, anB, BSZ, H);
  }
  sim_topk_kernel<<<BSZ / 16, 32, 0, stream>>>(anA, anB, tkv, tki);
  finalize_kernel<<<(BSZ + TB - 1) / TB, TB, 0, stream>>>(tkv, tki, y, plc, out_final);
}
